// VQ_16836271800400
// MI455X (gfx1250) — matrix operand order repaired and checked by reading, not compiled, not run on hardware
//
#include <hip/hip_runtime.h>
#include <hip/hip_fp16.h>

typedef __attribute__((ext_vector_type(16))) _Float16 v16h;
typedef __attribute__((ext_vector_type(8)))  float    v8f;

#define NB   16
#define CCH  256
#define HW_  4096
#define KA   512
#define EPSF 1e-12f
#define INV_TEMP (1.0f/0.3f)

#define WMMA16(a, b, c) \
  __builtin_amdgcn_wmma_f32_16x16x32_f16(false, (a), false, (b), (short)0, (c), false, false)

// ---------------- fragment loaders (CDNA5 16-bit WMMA layouts) ----------------
// A 16x32 f16: lane L holds row m=L&15; k pairs at halfs (v<4 ? 2v : 16+2(v-4)) + 8*(L>>4)
__device__ __forceinline__ v16h frag_a(const _Float16* row, int h) {
  union { v16h v; unsigned u[8]; } f;
  const unsigned* r = (const unsigned*)row;
#pragma unroll
  for (int i = 0; i < 8; ++i) f.u[i] = r[(i < 4 ? i : i + 4) + 4 * h];
  return f.v;
}
// B 32x16 f16 (stored as B^T rows, k contiguous): lane L holds col n=L&15;
// k pairs at halfs (v<4 ? 2v : 16+2(v-4)) + 8*(L>>4)
__device__ __forceinline__ v16h frag_b(const _Float16* row, int h) {
  union { v16h v; unsigned u[8]; } f;
  const unsigned* r = (const unsigned*)row;
#pragma unroll
  for (int i = 0; i < 8; ++i) f.u[i] = r[(i < 4 ? i : i + 4) + 4 * h];
  return f.v;
}

// ---------------- 1) channel-norm reciprocal per (n, p) ----------------
__global__ void __launch_bounds__(256) rnorm_kernel(const float* __restrict__ x,
                                                    float* __restrict__ rn) {
  int idx = blockIdx.x * 256 + threadIdx.x;     // n*HW + p
  int n = idx >> 12, p = idx & (HW_ - 1);
  const float* xp = x + (size_t)n * CCH * HW_ + p;
  float s = 0.f;
#pragma unroll 8
  for (int c = 0; c < CCH; ++c) { float v = xp[(size_t)c * HW_]; s += v * v; }
  rn[idx] = 1.0f / fmaxf(sqrtf(s), EPSF);
}

// ---------------- 2) transpose + normalize: Xh[n][p][c] = f16(l2_x) ----------------
__global__ void __launch_bounds__(1024) xpose_kernel(const float* __restrict__ x,
                                                     const float* __restrict__ rn,
                                                     _Float16* __restrict__ Xh) {
  __shared__ float tile[32][33];
  int n = blockIdx.z, p0 = blockIdx.x * 32, c0 = blockIdx.y * 32;
  int tx = threadIdx.x, ty = threadIdx.y;
  tile[ty][tx] = x[((size_t)(n * CCH + c0 + ty)) * HW_ + p0 + tx];
  __syncthreads();
  float r = rn[n * HW_ + p0 + ty];
  Xh[((size_t)(n * HW_ + p0 + ty)) * CCH + c0 + tx] = (_Float16)(tile[tx][ty] * r);
}

// ---------------- 3) gather anchors ----------------
__global__ void __launch_bounds__(256) gather_kernel(const _Float16* __restrict__ Xh,
                                                     const int* __restrict__ idx,
                                                     _Float16* __restrict__ sxT,
                                                     _Float16* __restrict__ sxM) {
  int j = blockIdx.x, n = blockIdx.y, c = threadIdx.x;
  int p = idx[j];
  _Float16 v = Xh[((size_t)(n * HW_ + p)) * CCH + c];
  sxT[((size_t)(n * KA + j)) * CCH + c] = v;   // [n][j][c] : GEMM1 B^T
  sxM[((size_t)(n * CCH + c)) * KA + j] = v;   // [n][c][j] : GEMM2 A
}

// ---------------- 4) GEMM1: A = exp( (Xh . sxT^T) / 0.3 ), 128x64 block tile ----------------
__global__ void __launch_bounds__(256) gemm1_kernel(const _Float16* __restrict__ Xh,
                                                    const _Float16* __restrict__ sxT,
                                                    float* __restrict__ Abuf) {
  __shared__ _Float16 As[2][128][40];
  __shared__ _Float16 Bs[2][64][40];
  int n = blockIdx.z, p0 = blockIdx.x * 128, j0 = blockIdx.y * 64;
  int t = threadIdx.x, wave = t >> 5, lane = t & 31;
  int rb = wave & 3, cb = wave >> 2;          // 4 row-blocks (32) x 2 col-blocks (32)
  int h = lane >> 4, ln = lane & 15;
  int lrow = t >> 2, loff = (t & 3) * 8;      // chunk-of-8-halfs decomposition
  const _Float16* Arow = Xh  + ((size_t)n * HW_ + p0) * CCH;
  const _Float16* Brow = sxT + ((size_t)n * KA  + j0) * CCH;
  v8f acc00 = {}, acc01 = {}, acc10 = {}, acc11 = {};

  float4 ra0, ra1, rb0;
  ra0 = *(const float4*)(Arow + (size_t)lrow * CCH + loff);
  ra1 = *(const float4*)(Arow + (size_t)(lrow + 64) * CCH + loff);
  rb0 = *(const float4*)(Brow + (size_t)lrow * CCH + loff);
  *(float4*)&As[0][lrow][loff]      = ra0;
  *(float4*)&As[0][lrow + 64][loff] = ra1;
  *(float4*)&Bs[0][lrow][loff]      = rb0;
  __syncthreads();

  const int S = CCH / 32;
#pragma unroll
  for (int s = 0; s < S; ++s) {
    int cur = s & 1;
    if (s + 1 < S) {
      int kc = (s + 1) * 32;
      ra0 = *(const float4*)(Arow + (size_t)lrow * CCH + kc + loff);
      ra1 = *(const float4*)(Arow + (size_t)(lrow + 64) * CCH + kc + loff);
      rb0 = *(const float4*)(Brow + (size_t)lrow * CCH + kc + loff);
    }
    v16h fa0 = frag_a(&As[cur][rb * 32 + ln][0], h);
    v16h fa1 = frag_a(&As[cur][rb * 32 + 16 + ln][0], h);
    v16h fb0 = frag_b(&Bs[cur][cb * 32 + ln][0], h);
    v16h fb1 = frag_b(&Bs[cur][cb * 32 + 16 + ln][0], h);
    acc00 = WMMA16(fa0, fb0, acc00);
    acc01 = WMMA16(fa0, fb1, acc01);
    acc10 = WMMA16(fa1, fb0, acc10);
    acc11 = WMMA16(fa1, fb1, acc11);
    if (s + 1 < S) {
      int nxt = 1 - cur;
      *(float4*)&As[nxt][lrow][loff]      = ra0;
      *(float4*)&As[nxt][lrow + 64][loff] = ra1;
      *(float4*)&Bs[nxt][lrow][loff]      = rb0;
      __syncthreads();
    }
  }

  size_t base = ((size_t)n * HW_ + p0 + rb * 32 + h * 8) * KA + j0 + cb * 32 + ln;
#pragma unroll
  for (int v = 0; v < 8; ++v) {
    Abuf[base + (size_t)v * KA]                       = __expf(acc00[v] * INV_TEMP);
    Abuf[base + (size_t)v * KA + 16]                  = __expf(acc01[v] * INV_TEMP);
    Abuf[base + (size_t)(v + 16) * KA]                = __expf(acc10[v] * INV_TEMP);
    Abuf[base + (size_t)(v + 16) * KA + 16]           = __expf(acc11[v] * INV_TEMP);
  }
}

// ---------------- 5a) column sums of A (atomics into cs[512]) ----------------
__global__ void __launch_bounds__(256) colsum_kernel(const float* __restrict__ Abuf,
                                                     float* __restrict__ cs) {
  int t = threadIdx.x;
  const float* p = Abuf + (size_t)blockIdx.x * 256 * KA + t;
  float s0 = 0.f, s1 = 0.f;
  for (int i = 0; i < 256; ++i) { s0 += p[0]; s1 += p[256]; p += KA; }
  atomicAdd(&cs[t], s0);
  atomicAdd(&cs[t + 256], s1);
}

// ---------------- 5b) fused column-divide + row-normalize ----------------
__global__ void __launch_bounds__(256) rownorm_kernel(float* __restrict__ Abuf,
                                                      const float* __restrict__ cs) {
  __shared__ float ws[8];
  int t = threadIdx.x;
  float c0 = 1.0f / fmaxf(cs[t], EPSF);
  float c1 = 1.0f / fmaxf(cs[t + 256], EPSF);
  size_t r0 = (size_t)blockIdx.x * 64;
  for (int i = 0; i < 64; ++i) {
    float* row = Abuf + (r0 + i) * KA;
    float v0 = row[t] * c0, v1 = row[t + 256] * c1;
    float s = v0 + v1;
#pragma unroll
    for (int off = 16; off; off >>= 1) s += __shfl_xor(s, off, 32);
    if ((t & 31) == 0) ws[t >> 5] = s;
    __syncthreads();
    float tot = ws[0] + ws[1] + ws[2] + ws[3] + ws[4] + ws[5] + ws[6] + ws[7];
    float inv = 1.0f / fmaxf(tot, EPSF);
    row[t] = v0 * inv;
    row[t + 256] = v1 * inv;
    __syncthreads();
  }
}

// ---------------- 6) GEMM2: out[n][c][p] = sum_k sxM[n][c][k] * A[n*HW+p][k] ----------------
__global__ void __launch_bounds__(256) gemm2_kernel(const _Float16* __restrict__ sxM,
                                                    const float* __restrict__ Abuf,
                                                    float* __restrict__ out) {
  __shared__ _Float16 As[2][128][40];
  __shared__ _Float16 Bs[2][64][40];
  int n = blockIdx.z, p0 = blockIdx.x * 64, c0 = blockIdx.y * 128;
  int t = threadIdx.x, wave = t >> 5, lane = t & 31;
  int rb = wave & 3, cb = wave >> 2;
  int h = lane >> 4, ln = lane & 15;
  int lrow = t >> 2, loff = (t & 3) * 8;
  const _Float16* Abase = sxM  + ((size_t)n * CCH + c0) * KA;
  const float*    Bbase = Abuf + ((size_t)n * HW_ + p0) * KA;
  v8f acc00 = {}, acc01 = {}, acc10 = {}, acc11 = {};

  float4 ra0, ra1, q0, q1;
  ra0 = *(const float4*)(Abase + (size_t)lrow * KA + loff);
  ra1 = *(const float4*)(Abase + (size_t)(lrow + 64) * KA + loff);
  q0  = *(const float4*)(Bbase + (size_t)lrow * KA + loff);
  q1  = *(const float4*)(Bbase + (size_t)lrow * KA + loff + 4);
  *(float4*)&As[0][lrow][loff]      = ra0;
  *(float4*)&As[0][lrow + 64][loff] = ra1;
  {
    _Float16* bs = &Bs[0][lrow][loff];
    bs[0] = (_Float16)q0.x; bs[1] = (_Float16)q0.y; bs[2] = (_Float16)q0.z; bs[3] = (_Float16)q0.w;
    bs[4] = (_Float16)q1.x; bs[5] = (_Float16)q1.y; bs[6] = (_Float16)q1.z; bs[7] = (_Float16)q1.w;
  }
  __syncthreads();

  const int S = KA / 32;
#pragma unroll
  for (int s = 0; s < S; ++s) {
    int cur = s & 1;
    if (s + 1 < S) {
      int kc = (s + 1) * 32;
      ra0 = *(const float4*)(Abase + (size_t)lrow * KA + kc + loff);
      ra1 = *(const float4*)(Abase + (size_t)(lrow + 64) * KA + kc + loff);
      q0  = *(const float4*)(Bbase + (size_t)lrow * KA + kc + loff);
      q1  = *(const float4*)(Bbase + (size_t)lrow * KA + kc + loff + 4);
    }
    v16h fa0 = frag_a(&As[cur][rb * 32 + ln][0], h);
    v16h fa1 = frag_a(&As[cur][rb * 32 + 16 + ln][0], h);
    v16h fb0 = frag_b(&Bs[cur][cb * 32 + ln][0], h);
    v16h fb1 = frag_b(&Bs[cur][cb * 32 + 16 + ln][0], h);
    acc00 = WMMA16(fa0, fb0, acc00);
    acc01 = WMMA16(fa0, fb1, acc01);
    acc10 = WMMA16(fa1, fb0, acc10);
    acc11 = WMMA16(fa1, fb1, acc11);
    if (s + 1 < S) {
      int nxt = 1 - cur;
      *(float4*)&As[nxt][lrow][loff]      = ra0;
      *(float4*)&As[nxt][lrow + 64][loff] = ra1;
      _Float16* bs = &Bs[nxt][lrow][loff];
      bs[0] = (_Float16)q0.x; bs[1] = (_Float16)q0.y; bs[2] = (_Float16)q0.z; bs[3] = (_Float16)q0.w;
      bs[4] = (_Float16)q1.x; bs[5] = (_Float16)q1.y; bs[6] = (_Float16)q1.z; bs[7] = (_Float16)q1.w;
      __syncthreads();
    }
  }

  size_t obase = ((size_t)n * CCH + c0 + rb * 32 + h * 8) * HW_ + p0 + cb * 32 + ln;
#pragma unroll
  for (int v = 0; v < 8; ++v) {
    out[obase + (size_t)v * HW_]             = acc00[v];
    out[obase + (size_t)v * HW_ + 16]        = acc01[v];
    out[obase + (size_t)(v + 16) * HW_]      = acc10[v];
    out[obase + (size_t)(v + 16) * HW_ + 16] = acc11[v];
  }
}

// ---------------- host-side launch ----------------
extern "C" void kernel_launch(void* const* d_in, const int* in_sizes, int n_in,
                              void* d_out, int out_size, void* d_ws, size_t ws_size,
                              hipStream_t stream) {
  (void)in_sizes; (void)n_in; (void)out_size; (void)ws_size;
  const float* x        = (const float*)d_in[0];
  const int*   some_idx = (const int*)d_in[1];
  float*       out      = (float*)d_out;

  char* ws = (char*)d_ws;
  size_t off = 0;
  auto carve = [&](size_t bytes) -> void* {
    void* p = (void*)(ws + off);
    off += (bytes + 255) & ~(size_t)255;
    return p;
  };
  _Float16* Xh   = (_Float16*)carve((size_t)NB * HW_ * CCH * sizeof(_Float16)); // 32 MB
  _Float16* sxT  = (_Float16*)carve((size_t)NB * KA  * CCH * sizeof(_Float16)); //  4 MB
  _Float16* sxM  = (_Float16*)carve((size_t)NB * CCH * KA  * sizeof(_Float16)); //  4 MB
  float*    rn   = (float*)   carve((size_t)NB * HW_ * sizeof(float));          // 256 KB
  float*    cs   = (float*)   carve((size_t)KA * sizeof(float));                //  2 KB
  float*    Abuf = (float*)   carve((size_t)NB * HW_ * KA * sizeof(float));     // 128 MB

  rnorm_kernel<<<(NB * HW_) / 256, 256, 0, stream>>>(x, rn);
  xpose_kernel<<<dim3(HW_ / 32, CCH / 32, NB), dim3(32, 32), 0, stream>>>(x, rn, Xh);
  gather_kernel<<<dim3(KA, NB), 256, 0, stream>>>(Xh, some_idx, sxT, sxM);
  gemm1_kernel<<<dim3(HW_ / 128, KA / 64, NB), 256, 0, stream>>>(Xh, sxT, Abuf);
  for (int it = 0; it < 4; ++it) {
    hipMemsetAsync(cs, 0, KA * sizeof(float), stream);
    colsum_kernel<<<(NB * HW_) / 256, 256, 0, stream>>>(Abuf, cs);
    rownorm_kernel<<<(NB * HW_) / 64, 256, 0, stream>>>(Abuf, cs);
  }
  gemm2_kernel<<<dim3(HW_ / 64, CCH / 128, NB), 256, 0, stream>>>(sxM, Abuf, out);
}
